// SelfAttentionHead_88390426952036
// MI455X (gfx1250) — matrix operand order repaired and checked by reading, not compiled, not run on hardware
//
#include <hip/hip_runtime.h>
#include <hip/hip_bf16.h>

// Self-attention head, fused WMMA implementation for gfx1250 (wave32).
// B=256, T=256, D=384, HD=64. fp32 in/out, f16 WMMA math with f32 accum.

#define B_DIM 256
#define T_DIM 256
#define D_DIM 384
#define HD_DIM 64

typedef __attribute__((ext_vector_type(16))) _Float16 v16h;
typedef __attribute__((ext_vector_type(8)))  _Float16 v8h;
typedef __attribute__((ext_vector_type(8)))  float    v8f;
typedef _Float16 half_t;

__device__ __forceinline__ v8f wmma_f16(const v16h& a, const v16h& b, const v8f& c) {
    return __builtin_amdgcn_wmma_f32_16x16x32_f16(
        /*neg_a=*/false, a, /*neg_b=*/false, b,
        /*c_mod=*/(short)0, c, /*reuse_a=*/false, /*reuse_b=*/false);
}

// ---------------------------------------------------------------------------
// Kernel 0: one-shot weight conversion fp32 -> f16 (Wq|Wk|Wv, 73728 elements).
// Done once so the 4096 projection waves load f16 fragments with NO cvt.
// ---------------------------------------------------------------------------
__global__ __launch_bounds__(256)
void w_to_f16_kernel(const float* __restrict__ Wq,
                     const float* __restrict__ Wk,
                     const float* __restrict__ Wv,
                     half_t* __restrict__ Wh) {
    const int n   = HD_DIM * D_DIM;          // 24576 per weight
    const int idx = blockIdx.x * 256 + threadIdx.x;   // 0 .. 3n-1
    const float* src = (idx < n) ? Wq : (idx < 2 * n) ? Wk : Wv;
    const int off    = (idx < n) ? idx : (idx < 2 * n) ? idx - n : idx - 2 * n;
    Wh[idx] = (half_t)src[off];
}

// ---------------------------------------------------------------------------
// Kernel 1: Q/K/V projections.  One wave (32 threads) per 16-row tile of x.
// Single pass over x: A fragment loaded once per K-step (fp32 -> f16 cvt),
// weight B-fragments are direct 32-byte v16h loads (layout == memory order),
// 12 live accumulators (3 weights x 4 N-tiles), fully unrolled.
//   Q[b,t,h], K[b,t,h] row-major f16;  Vt[b,h,t] (transposed) f16.
// A-fragment (16x32 f16): lane m = lane&15, half = lane>>4,
//   halves [0..7]  = K (s*32 + half*16     + 0..7)
//   halves [8..15] = K (s*32 + half*16 + 8 + 0..7)
// B-fragment (32x16 f16): lane n = lane&15,
//   halves [0..15] = K (s*32 + half*16 + 0..15)  -> contiguous in memory
// C/D (16x16 f32): VGPR r -> row m = r + half*8, col n = lane&15.
// ---------------------------------------------------------------------------
__global__ __launch_bounds__(32)
void attn_proj_kernel(const float* __restrict__ x,
                      const half_t* __restrict__ Wh,   // f16 Wq|Wk|Wv
                      half_t* __restrict__ Q,
                      half_t* __restrict__ K,
                      half_t* __restrict__ Vt) {
    const int tile  = blockIdx.x;            // 0 .. B*T/16-1
    const int b     = tile >> 4;             // 16 tiles per batch (T=256)
    const int t0    = (tile & 15) * 16;
    const int lane  = threadIdx.x;
    const int mn    = lane & 15;
    const int half_ = lane >> 4;

    const float* xrow = x + ((size_t)b * T_DIM + t0 + mn) * D_DIM;
    const v8f zf = {0.f, 0.f, 0.f, 0.f, 0.f, 0.f, 0.f, 0.f};

    // acc[w][nt]: w=0 Q, w=1 K, w=2 V.  96 VGPRs of f32 accumulators.
    v8f acc[3][4];
    #pragma unroll
    for (int w = 0; w < 3; ++w)
        #pragma unroll
        for (int nt = 0; nt < 4; ++nt) acc[w][nt] = zf;

    for (int s = 0; s < D_DIM / 32; ++s) {
        // ---- A fragment from x: 4x b128 fp32 loads + 8 cvt_pk ----
        const int base0 = s * 32 + half_ * 16;       // halves [0..7]
        const int base1 = s * 32 + half_ * 16 + 8;   // halves [8..15]
        const float4 fa0 = *(const float4*)(xrow + base0);
        const float4 fa1 = *(const float4*)(xrow + base0 + 4);
        const float4 fa2 = *(const float4*)(xrow + base1);
        const float4 fa3 = *(const float4*)(xrow + base1 + 4);
        v16h a;
        a[0]  = (_Float16)fa0.x; a[1]  = (_Float16)fa0.y;
        a[2]  = (_Float16)fa0.z; a[3]  = (_Float16)fa0.w;
        a[4]  = (_Float16)fa1.x; a[5]  = (_Float16)fa1.y;
        a[6]  = (_Float16)fa1.z; a[7]  = (_Float16)fa1.w;
        a[8]  = (_Float16)fa2.x; a[9]  = (_Float16)fa2.y;
        a[10] = (_Float16)fa2.z; a[11] = (_Float16)fa2.w;
        a[12] = (_Float16)fa3.x; a[13] = (_Float16)fa3.y;
        a[14] = (_Float16)fa3.z; a[15] = (_Float16)fa3.w;

        #pragma unroll
        for (int w = 0; w < 3; ++w) {
            const half_t* Wbase = Wh + (size_t)w * HD_DIM * D_DIM;
            #pragma unroll
            for (int nt = 0; nt < 4; ++nt) {
                // B fragment == 16 contiguous f16: single 32-byte load.
                const v16h bf = *(const v16h*)(Wbase +
                    (size_t)(nt * 16 + mn) * D_DIM + s * 32 + half_ * 16);
                acc[w][nt] = wmma_f16(a, bf, acc[w][nt]);
            }
        }
    }

    // ---- branch-free stores: w is compile-time constant in each block ----
    #pragma unroll
    for (int nt = 0; nt < 4; ++nt) {
        #pragma unroll
        for (int r = 0; r < 8; ++r) {
            const int m = r + half_ * 8;
            const int n = nt * 16 + mn;
            Q[((size_t)b * T_DIM + t0 + m) * HD_DIM + n] = (half_t)acc[0][nt][r];
        }
    }
    #pragma unroll
    for (int nt = 0; nt < 4; ++nt) {
        #pragma unroll
        for (int r = 0; r < 8; ++r) {
            const int m = r + half_ * 8;
            const int n = nt * 16 + mn;
            K[((size_t)b * T_DIM + t0 + m) * HD_DIM + n] = (half_t)acc[1][nt][r];
        }
    }
    #pragma unroll
    for (int nt = 0; nt < 4; ++nt) {
        #pragma unroll
        for (int r = 0; r < 8; ++r) {
            const int m = r + half_ * 8;
            const int n = nt * 16 + mn;
            Vt[((size_t)b * HD_DIM + n) * T_DIM + t0 + m] = (half_t)acc[2][nt][r];
        }
    }
}

// ---------------------------------------------------------------------------
// Kernel 2: causal flash attention.  One wave per (batch, 16-query tile).
// Streams 32-key chunks: scores = Q(16x64) x K^T (2 C-tiles, 2 K-steps each),
// online softmax with row reductions over 16-lane halves, P through LDS
// (C-layout -> A-layout transpose), O += P(16x32) x V(32x64) (4 WMMAs).
// All fragments are vector loads (b128): layouts were chosen so fragment
// halves are memory-contiguous.
// ---------------------------------------------------------------------------
__global__ __launch_bounds__(32)
void attn_flash_kernel(const half_t* __restrict__ Q,
                       const half_t* __restrict__ K,
                       const half_t* __restrict__ Vt,
                       float* __restrict__ out) {
    __shared__ half_t Plds[16 * 32];

    const int tile  = blockIdx.x;
    const int b     = tile >> 4;
    const int i     = tile & 15;             // query tile within batch
    const int t0    = i * 16;
    const int lane  = threadIdx.x;
    const int mn    = lane & 15;
    const int half_ = lane >> 4;

    const v8f zf = {0.f, 0.f, 0.f, 0.f, 0.f, 0.f, 0.f, 0.f};

    // ---- Q A-fragments, K-dim 64 = 2 chunks of 32 (loaded once) ----
    const half_t* qrow = Q + ((size_t)b * T_DIM + t0 + mn) * HD_DIM;
    v16h qa[2];
    #pragma unroll
    for (int s = 0; s < 2; ++s) {
        const v8h lo = *(const v8h*)(qrow + s * 32 + half_ * 16);
        const v8h hi = *(const v8h*)(qrow + s * 32 + half_ * 16 + 8);
        #pragma unroll
        for (int e = 0; e < 8; ++e) { qa[s][e] = lo[e]; qa[s][8 + e] = hi[e]; }
    }

    v8f   oacc[4] = {zf, zf, zf, zf};        // 16 x 64 output accumulator
    float m_run[8], l_run[8];
    #pragma unroll
    for (int r = 0; r < 8; ++r) { m_run[r] = -1e30f; l_run[r] = 0.f; }

    const float scale = 0.125f;              // 1/sqrt(64)
    const int nchunk = (i + 2) >> 1;         // ceil(16*(i+1)/32): causal stop

    for (int j = 0; j < nchunk; ++j) {
        const int kbase = j * 32;

        // ---- scores: 2 C-tiles (16 keys each), K-dim 64 ----
        v8f sc[2] = {zf, zf};
        #pragma unroll
        for (int c = 0; c < 2; ++c) {
            const half_t* krow =
                K + ((size_t)b * T_DIM + kbase + c * 16 + mn) * HD_DIM;
            #pragma unroll
            for (int s = 0; s < 2; ++s) {
                const v16h kb = *(const v16h*)(krow + s * 32 + half_ * 16);
                sc[c] = wmma_f16(qa[s], kb, sc[c]);
            }
        }

        // ---- scale + causal mask (branch-free; EXEC stays all-1s) ----
        #pragma unroll
        for (int c = 0; c < 2; ++c) {
            const int kk = kbase + c * 16 + mn;
            #pragma unroll
            for (int r = 0; r < 8; ++r) {
                const int qg = t0 + r + half_ * 8;
                const float sv = sc[c][r] * scale;
                sc[c][r] = (kk > qg) ? -1e30f : sv;
            }
        }

        // ---- online softmax: per-row (16-lane half) reductions ----
        float alpha[8];
        #pragma unroll
        for (int r = 0; r < 8; ++r) {
            float mx = fmaxf(sc[0][r], sc[1][r]);
            for (int off = 1; off < 16; off <<= 1)
                mx = fmaxf(mx, __shfl_xor(mx, off, 32));
            const float mnew = fmaxf(m_run[r], mx);
            alpha[r] = __expf(m_run[r] - mnew);
            m_run[r] = mnew;
            const float p0 = __expf(sc[0][r] - mnew);
            const float p1 = __expf(sc[1][r] - mnew);
            sc[0][r] = p0;
            sc[1][r] = p1;
            float sum = p0 + p1;
            for (int off = 1; off < 16; off <<= 1)
                sum += __shfl_xor(sum, off, 32);
            l_run[r] = l_run[r] * alpha[r] + sum;
            #pragma unroll
            for (int nt = 0; nt < 4; ++nt) oacc[nt][r] *= alpha[r];
        }

        // ---- P: C-layout -> A-layout via LDS (16x32 f16) ----
        __syncthreads();
        #pragma unroll
        for (int c = 0; c < 2; ++c) {
            #pragma unroll
            for (int r = 0; r < 8; ++r) {
                Plds[(r + half_ * 8) * 32 + c * 16 + mn] = (half_t)sc[c][r];
            }
        }
        __syncthreads();
        v16h pa;
        {
            const v8h plo = *(const v8h*)(Plds + mn * 32 + half_ * 16);
            const v8h phi = *(const v8h*)(Plds + mn * 32 + half_ * 16 + 8);
            #pragma unroll
            for (int e = 0; e < 8; ++e) { pa[e] = plo[e]; pa[8 + e] = phi[e]; }
        }

        // ---- O += P(16x32) x V(32x64): 4 WMMAs, V^T layout -> b128 loads ----
        #pragma unroll
        for (int nt = 0; nt < 4; ++nt) {
            const half_t* vrow =
                Vt + ((size_t)b * HD_DIM + nt * 16 + mn) * T_DIM + kbase;
            const v16h vb = *(const v16h*)(vrow + half_ * 16);
            oacc[nt] = wmma_f16(pa, vb, oacc[nt]);
        }
    }

    // ---- normalize and store fp32 output ----
    float* orow = out + ((size_t)b * T_DIM + t0) * HD_DIM;
    #pragma unroll
    for (int r = 0; r < 8; ++r) {
        const float inv = 1.0f / l_run[r];
        const int m = r + half_ * 8;
        #pragma unroll
        for (int nt = 0; nt < 4; ++nt)
            orow[(size_t)m * HD_DIM + nt * 16 + mn] = oacc[nt][r] * inv;
    }
}

extern "C" void kernel_launch(void* const* d_in, const int* in_sizes, int n_in,
                              void* d_out, int out_size, void* d_ws, size_t ws_size,
                              hipStream_t stream) {
    const float* x  = (const float*)d_in[0];
    const float* Wq = (const float*)d_in[1];
    const float* Wk = (const float*)d_in[2];
    const float* Wv = (const float*)d_in[3];
    float* out = (float*)d_out;

    // Workspace: Q | K | Vt (each B*T*HD f16, 8 MB) | Wh (3*HD*D f16, 144 KB).
    const size_t qkv_elems = (size_t)B_DIM * T_DIM * HD_DIM;
    half_t* Q  = (half_t*)d_ws;
    half_t* K  = Q + qkv_elems;
    half_t* Vt = K + qkv_elems;
    half_t* Wh = Vt + qkv_elems;

    const int wtotal = 3 * HD_DIM * D_DIM;   // 73728 weight elements
    w_to_f16_kernel<<<dim3(wtotal / 256), dim3(256), 0, stream>>>(Wq, Wk, Wv, Wh);

    const int ntiles = B_DIM * T_DIM / 16;   // 4096 waves, one per 16-row tile
    attn_proj_kernel<<<dim3(ntiles), dim3(32), 0, stream>>>(x, Wh, Q, K, Vt);
    attn_flash_kernel<<<dim3(ntiles), dim3(32), 0, stream>>>(Q, K, Vt, out);
}
